// GQA_34711925686465
// MI455X (gfx1250) — matrix operand order repaired and checked by reading, not compiled, not run on hardware
//
#include <hip/hip_runtime.h>
#include <hip/hip_bf16.h>

// ---------------------------------------------------------------------------
// GQA attention block for gfx1250 (MI455X), bf16 WMMA everywhere.
//   B=2, T=2048, C=2048, Hq=32, Hkv=8, D=64, causal, scale = 1/8.
// Pipeline:
//   1) cast fp32 -> bf16 (x, wq, wk, wv, wo)
//   2) q/k/v projections : TDM double-buffered LDS-tiled WMMA GEMM (bf16 out)
//   3) flash-attention   : WMMA QK^T + DPP online softmax + WMMA PV (bf16 out)
//   4) out = o@wo^T      : same GEMM, fp32 out
// ---------------------------------------------------------------------------

typedef __attribute__((ext_vector_type(16))) __bf16        v16bf;
typedef __attribute__((ext_vector_type(8)))  float         v8f;
typedef __attribute__((ext_vector_type(4)))  unsigned int  u32x4;
typedef __attribute__((ext_vector_type(4)))  int           i32x4;
typedef __attribute__((ext_vector_type(8)))  int           i32x8;
typedef __attribute__((ext_vector_type(4)))  float         f32x4;

#if defined(__has_builtin)
#if __has_builtin(__builtin_amdgcn_tensor_load_to_lds) && \
    __has_builtin(__builtin_amdgcn_s_wait_tensorcnt)
#define HAVE_TDM 1
#endif
#endif
#ifndef HAVE_TDM
#define HAVE_TDM 0
#endif

__device__ __forceinline__ unsigned short f2bf(float x) {
    unsigned int u = __float_as_uint(x);
    u += 0x7FFFu + ((u >> 16) & 1u);          // round-to-nearest-even
    return (unsigned short)(u >> 16);
}

union Frag32B { u32x4 u[2]; v16bf v; };

__device__ __forceinline__ v16bf ld_frag2(const unsigned short* p0,
                                          const unsigned short* p1) {
    Frag32B t;
    t.u[0] = *(const u32x4*)p0;
    t.u[1] = *(const u32x4*)p1;
    return t.v;
}
// A-matrix 16x32 bf16 fragment: lane<16 -> K {0..15}; lane>=16 -> {16..31}
// caller passes p = row_base + (lane>=16 ? 16 : 0); reads p[0..15].
__device__ __forceinline__ v16bf ld_frag_a(const unsigned short* p) {
    return ld_frag2(p, p + 8);
}
// B-matrix 32x16 bf16 fragment: lane = column n; 16 contiguous K values.
__device__ __forceinline__ v16bf ld_frag_b(const unsigned short* p) {
    return ld_frag2(p, p + 8);
}

__device__ __forceinline__ v8f wmma_bf16(v16bf a, v16bf b, v8f c) {
    return __builtin_amdgcn_wmma_f32_16x16x32_bf16(false, a, false, b,
                                                   (short)0, c, false, false);
}

// ---- 16-lane row reductions via DPP row rotations (pure VALU, no LDS) ----
#define DPP_ROR_F(x, n)                                                       \
    __int_as_float(__builtin_amdgcn_mov_dpp(__float_as_int(x), 0x120 + (n),   \
                                            0xf, 0xf, false))
__device__ __forceinline__ float row16_max(float x) {
    x = fmaxf(x, DPP_ROR_F(x, 1));
    x = fmaxf(x, DPP_ROR_F(x, 2));
    x = fmaxf(x, DPP_ROR_F(x, 4));
    x = fmaxf(x, DPP_ROR_F(x, 8));
    return x;
}
__device__ __forceinline__ float row16_sum(float x) {
    x += DPP_ROR_F(x, 1);
    x += DPP_ROR_F(x, 2);
    x += DPP_ROR_F(x, 4);
    x += DPP_ROR_F(x, 8);
    return x;
}

// ---------------------------------------------------------------------------
// Tensor Data Mover: 2-D bf16 tile load, global -> LDS (ISA ch.8 D# layout).
//   tile_rows x tile_cols elements, tensor row length = row_stride elements.
// ---------------------------------------------------------------------------
#if HAVE_TDM
__device__ __forceinline__ void tdm_load_2d(const unsigned short* gsrc,
                                            const unsigned short* lds_dst,
                                            int tile_rows, int tile_cols,
                                            int tensor_rows,
                                            long long row_stride) {
    unsigned long long ga = (unsigned long long)(size_t)gsrc;
    unsigned int la = (unsigned int)(size_t)lds_dst;  // LDS aperture: addr[31:0]
    unsigned int td0 = (unsigned int)row_stride;      // tensor_dim0 (elems)
    unsigned int td1 = (unsigned int)tensor_rows;     // tensor_dim1
    unsigned long long s0 = (unsigned long long)row_stride;

    u32x4 g0;
    g0.x = 1u;                                        // count=1, user D#
    g0.y = la;                                        // lds_addr
    g0.z = (unsigned int)(ga & 0xffffffffu);          // global_addr[31:0]
    g0.w = (unsigned int)((ga >> 32) & 0x1ffffffu)    // global_addr[56:32]
         | (2u << 30);                                // type = 2 ("image")

    i32x8 g1;
    g1[0] = (int)(1u << 16);                          // data_size = 2 bytes
    g1[1] = (int)((td0 & 0xffffu) << 16);             // tensor_dim0 lo
    g1[2] = (int)(((td0 >> 16) & 0xffffu) | ((td1 & 0xffffu) << 16));
    g1[3] = (int)(((td1 >> 16) & 0xffffu) |
                  (((unsigned)tile_cols & 0xffffu) << 16));   // tile_dim0
    g1[4] = (int)((unsigned)tile_rows & 0xffffu);             // tile_dim1
    g1[5] = (int)(unsigned int)(s0 & 0xffffffffu);            // dim0_stride lo
    g1[6] = (int)(unsigned int)((s0 >> 32) & 0xffffu);        // dim0_stride hi
    g1[7] = 0;

    i32x4 g2 = {0, 0, 0, 0};
    i32x4 g3 = {0, 0, 0, 0};
#if __clang_major__ >= 23
    i32x8 g4 = {0, 0, 0, 0, 0, 0, 0, 0};
    __builtin_amdgcn_tensor_load_to_lds(g0, g1, g2, g3, g4, 0);
#else
    __builtin_amdgcn_tensor_load_to_lds(g0, g1, g2, g3, 0);
#endif
}
#endif

// ---------------------------------------------------------------------------
// fp32 -> bf16 cast, 4 elements per thread (b128 load, b64 store)
// ---------------------------------------------------------------------------
__global__ __launch_bounds__(256) void cast_bf16_kernel(
    const float* __restrict__ in, unsigned short* __restrict__ out, int n4) {
    int i = blockIdx.x * blockDim.x + threadIdx.x;
    if (i >= n4) return;
    f32x4 f = ((const f32x4*)in)[i];
    unsigned long long p = (unsigned long long)f2bf(f.x)
                         | ((unsigned long long)f2bf(f.y) << 16)
                         | ((unsigned long long)f2bf(f.z) << 32)
                         | ((unsigned long long)f2bf(f.w) << 48);
    ((unsigned long long*)out)[i] = p;
}

// ---------------------------------------------------------------------------
// Tiled WMMA GEMM:  C[m,n] = sum_k A[m,k] * B[n,k]   (A: MxK bf16, B: NxK bf16)
// 256 threads = 8 waves, block tile 128x128, K-step 64 (16 WMMAs per stage).
// TDM path: wave 0 DMAs stage s+1 into the ping-pong buffer during compute.
// ---------------------------------------------------------------------------
template <bool OUT_BF16>
__global__ __launch_bounds__(256) void gemm_wmma_kernel(
    const unsigned short* __restrict__ A, const unsigned short* __restrict__ B,
    void* __restrict__ C, int M, int N, int K) {
    __shared__ unsigned short As[2][128 * 64];    // 32 KB
    __shared__ unsigned short Bs[2][128 * 64];    // 32 KB

    const int tid  = threadIdx.x;
    const int lane = tid & 31;
    const int wid  = tid >> 5;
    const int wm   = wid >> 2;          // 0..1  (64 rows each)
    const int wn   = wid & 3;           // 0..3  (32 cols each)
    const int half = lane >> 4;
    const int l16  = lane & 15;
    const int m0   = blockIdx.y * 128;
    const int n0   = blockIdx.x * 128;

    v8f acc[4][2];
    #pragma unroll
    for (int mf = 0; mf < 4; ++mf)
        #pragma unroll
        for (int nf = 0; nf < 2; ++nf) acc[mf][nf] = (v8f)0.0f;

    const int nstage = K >> 6;          // K / 64
    int buf = 0;

#if HAVE_TDM
    if (wid == 0) {
        tdm_load_2d(&A[(size_t)m0 * K], &As[0][0], 128, 64, M, K);
        tdm_load_2d(&B[(size_t)n0 * K], &Bs[0][0], 128, 64, N, K);
        __builtin_amdgcn_s_wait_tensorcnt(0);
    }
    __syncthreads();
#endif

    for (int s = 0; s < nstage; ++s) {
        const int k0 = s << 6;
#if HAVE_TDM
        if (wid == 0 && s + 1 < nstage) {   // DMA next stage while computing
            tdm_load_2d(&A[(size_t)m0 * K + k0 + 64], &As[buf ^ 1][0], 128, 64, M, K);
            tdm_load_2d(&B[(size_t)n0 * K + k0 + 64], &Bs[buf ^ 1][0], 128, 64, N, K);
        }
#else
        // synchronous staging: 1024 b128 per operand, 4 each per thread
        #pragma unroll
        for (int r = 0; r < 4; ++r) {
            int q   = tid + r * 256;
            int row = q >> 3;
            int kq  = (q & 7) * 8;
            *(u32x4*)&As[0][row * 64 + kq] =
                *(const u32x4*)&A[(size_t)(m0 + row) * K + k0 + kq];
            *(u32x4*)&Bs[0][row * 64 + kq] =
                *(const u32x4*)&B[(size_t)(n0 + row) * K + k0 + kq];
            if (k0 + 64 < K) {
                __builtin_prefetch(&A[(size_t)(m0 + row) * K + k0 + 64 + kq], 0, 0);
                __builtin_prefetch(&B[(size_t)(n0 + row) * K + k0 + 64 + kq], 0, 0);
            }
        }
        __syncthreads();
#endif
        const unsigned short* as = &As[buf][0];
        const unsigned short* bs = &Bs[buf][0];
        #pragma unroll
        for (int kc = 0; kc < 2; ++kc) {
            const int kc32 = kc * 32;
            v16bf af[4], bfr[2];
            #pragma unroll
            for (int mf = 0; mf < 4; ++mf) {
                int m = wm * 64 + mf * 16 + l16;
                af[mf] = ld_frag_a(&as[m * 64 + kc32 + half * 16]);
            }
            #pragma unroll
            for (int nf = 0; nf < 2; ++nf) {
                int n = wn * 32 + nf * 16 + l16;
                bfr[nf] = ld_frag_b(&bs[n * 64 + kc32 + half * 16]);
            }
            #pragma unroll
            for (int mf = 0; mf < 4; ++mf)
                #pragma unroll
                for (int nf = 0; nf < 2; ++nf)
                    acc[mf][nf] = wmma_bf16(af[mf], bfr[nf], acc[mf][nf]);
        }
#if HAVE_TDM
        if (wid == 0 && s + 1 < nstage) __builtin_amdgcn_s_wait_tensorcnt(0);
        __syncthreads();
        buf ^= 1;
#else
        __syncthreads();
#endif
    }

    // epilogue: C/D layout -> lane l16 = column, VGPR i = row (i + 8*half)
    #pragma unroll
    for (int mf = 0; mf < 4; ++mf)
        #pragma unroll
        for (int nf = 0; nf < 2; ++nf) {
            int n = n0 + wn * 32 + nf * 16 + l16;
            #pragma unroll
            for (int i = 0; i < 8; ++i) {
                int m = m0 + wm * 64 + mf * 16 + i + 8 * half;
                float vv = acc[mf][nf][i];
                if (OUT_BF16)
                    ((unsigned short*)C)[(size_t)m * N + n] = f2bf(vv);
                else
                    ((float*)C)[(size_t)m * N + n] = vv;
            }
        }
}

// ---------------------------------------------------------------------------
// Flash attention (causal, GQA). Grid = (T/64, Hq, B), block = 128 (4 waves).
// Each wave owns 16 query rows; key blocks of 32; D = 64.
//   q: [B*T, 2048] bf16 (col = h*64+d)     k,v: [B*T, 512] bf16 (col = kvh*64+d)
//   o: [B*T, 2048] bf16
// ---------------------------------------------------------------------------
__global__ __launch_bounds__(128) void attn_wmma_kernel(
    const unsigned short* __restrict__ q, const unsigned short* __restrict__ k,
    const unsigned short* __restrict__ v, unsigned short* __restrict__ o) {
    __shared__ unsigned short Vt[64 * 32];        // [dim][key] transposed V tile
    __shared__ unsigned short Pb[4 * 16 * 32];    // per-wave P tile (16x32)

    const int tid  = threadIdx.x;
    const int lane = tid & 31;
    const int wvid = tid >> 5;
    const int half = lane >> 4;
    const int l16  = lane & 15;

    const int bq  = blockIdx.x;                   // query block (64 rows)
    const int h   = blockIdx.y;
    const int b   = blockIdx.z;
    const int kvh = h >> 2;                       // GQA: 4 q-heads per kv-head
    const int qw  = bq * 64 + wvid * 16;          // wave's first query row

    // Q A-fragments held in registers for the whole K loop
    const unsigned short* qp =
        q + ((size_t)(b * 2048 + qw + l16)) * 2048 + h * 64 + half * 16;
    v16bf aq0 = ld_frag_a(qp);        // d 0..31
    v16bf aq1 = ld_frag_a(qp + 32);   // d 32..63

    v8f o_acc[4];
    #pragma unroll
    for (int df = 0; df < 4; ++df) o_acc[df] = (v8f)0.0f;
    float rmax[8], rsum[8];
    #pragma unroll
    for (int i = 0; i < 8; ++i) { rmax[i] = -1e30f; rsum[i] = 0.0f; }

    const int nkb = bq * 2 + 2;                   // causal: keys <= block end
    for (int kb = 0; kb < nkb; ++kb) {
        const int kbase = kb * 32;

        // ---- stage V tile transposed into LDS: Vt[d][key] ----
        {
            int kp = tid & 15;                    // key pair 0..15
            int dg = tid >> 4;                    // dim group 0..7 (8 dims each)
            int key = kbase + kp * 2;
            const unsigned short* vp =
                v + ((size_t)(b * 2048 + key)) * 512 + kvh * 64 + dg * 8;
            union { u32x4 u; unsigned short s[8]; } r0, r1;
            r0.u = *(const u32x4*)vp;
            r1.u = *(const u32x4*)(vp + 512);
            #pragma unroll
            for (int j = 0; j < 8; ++j) {
                unsigned int pk = (unsigned int)r0.s[j] |
                                  ((unsigned int)r1.s[j] << 16);
                *(unsigned int*)&Vt[(dg * 8 + j) * 32 + kp * 2] = pk;
            }
        }
        __syncthreads();

        // ---- S = Q K^T (two 16x16 column tiles), scale + causal mask ----
        v8f s[2];
        #pragma unroll
        for (int nt = 0; nt < 2; ++nt) {
            int key = kbase + nt * 16 + l16;
            const unsigned short* kp_ =
                k + ((size_t)(b * 2048 + key)) * 512 + kvh * 64;
            v16bf bk0 = ld_frag_b(kp_ + half * 16);        // d 0..31
            v16bf bk1 = ld_frag_b(kp_ + 32 + half * 16);   // d 32..63
            v8f sv = (v8f)0.0f;
            sv = wmma_bf16(aq0, bk0, sv);
            sv = wmma_bf16(aq1, bk1, sv);
            #pragma unroll
            for (int i = 0; i < 8; ++i) {
                int qrow = qw + i + 8 * half;
                float x = sv[i] * 0.125f;                  // 1/sqrt(64)
                sv[i] = (key <= qrow) ? x : -1e30f;
            }
            s[nt] = sv;
        }

        // ---- online softmax: DPP row reductions, exp, rescale ----
        float fac[8];
        #pragma unroll
        for (int i = 0; i < 8; ++i) {
            float m  = row16_max(fmaxf(s[0][i], s[1][i]));
            float nm = fmaxf(rmax[i], m);
            fac[i]   = __expf(rmax[i] - nm);
            rmax[i]  = nm;
        }
        #pragma unroll
        for (int nt = 0; nt < 2; ++nt)
            #pragma unroll
            for (int i = 0; i < 8; ++i)
                s[nt][i] = __expf(s[nt][i] - rmax[i]);
        #pragma unroll
        for (int i = 0; i < 8; ++i) {
            float rs = row16_sum(s[0][i] + s[1][i]);
            rsum[i] = rsum[i] * fac[i] + rs;
        }
        #pragma unroll
        for (int df = 0; df < 4; ++df)
            #pragma unroll
            for (int i = 0; i < 8; ++i) o_acc[df][i] *= fac[i];

        // ---- P: C-layout -> A-layout via per-wave LDS tile ----
        unsigned short* pb = &Pb[wvid * 16 * 32];
        #pragma unroll
        for (int nt = 0; nt < 2; ++nt)
            #pragma unroll
            for (int i = 0; i < 8; ++i)
                pb[(i + 8 * half) * 32 + nt * 16 + l16] = f2bf(s[nt][i]);
        __syncthreads();
        v16bf ap = ld_frag_a(&pb[l16 * 32 + half * 16]);

        // ---- O += P V  (4 WMMAs over D=64) ----
        #pragma unroll
        for (int df = 0; df < 4; ++df) {
            v16bf bv = ld_frag_b(&Vt[(df * 16 + l16) * 32 + half * 16]);
            o_acc[df] = wmma_bf16(ap, bv, o_acc[df]);
        }
        __syncthreads();   // protect Vt/Pb before next iteration restages
    }

    // ---- normalize (v_rcp) and store bf16 ----
    #pragma unroll
    for (int df = 0; df < 4; ++df)
        #pragma unroll
        for (int i = 0; i < 8; ++i) {
            int qrow = qw + i + 8 * half;
            float val = o_acc[df][i] * __builtin_amdgcn_rcpf(rsum[i]);
            o[((size_t)(b * 2048 + qrow)) * 2048 + h * 64 + df * 16 + l16] =
                f2bf(val);
        }
}

// ---------------------------------------------------------------------------
// Host-side orchestration
// ---------------------------------------------------------------------------
extern "C" void kernel_launch(void* const* d_in, const int* in_sizes, int n_in,
                              void* d_out, int out_size, void* d_ws, size_t ws_size,
                              hipStream_t stream) {
    (void)in_sizes; (void)n_in; (void)out_size; (void)ws_size;
    const float* x  = (const float*)d_in[0];
    const float* wq = (const float*)d_in[1];
    const float* wk = (const float*)d_in[2];
    const float* wv = (const float*)d_in[3];
    const float* wo = (const float*)d_in[4];
    float* out = (float*)d_out;

    constexpr int Bz = 2, T = 2048, Cc = 2048, Hq = 32;
    constexpr int M = Bz * T;              // 4096
    constexpr int Nkv = 512;               // KV_HEADS * HEAD_DIM

    char* ws = (char*)d_ws;
    size_t off = 0;
    auto take = [&](size_t elems) {
        unsigned short* p = (unsigned short*)(ws + off);
        off = (off + elems * 2 + 255) & ~(size_t)255;
        return p;
    };
    unsigned short* xb  = take((size_t)M * Cc);     // 16 MB
    unsigned short* wqb = take((size_t)Cc * Cc);    //  8 MB
    unsigned short* wkb = take((size_t)Nkv * Cc);   //  2 MB
    unsigned short* wvb = take((size_t)Nkv * Cc);   //  2 MB
    unsigned short* wob = take((size_t)Cc * Cc);    //  8 MB
    unsigned short* qb  = take((size_t)M * Cc);     // 16 MB
    unsigned short* kb  = take((size_t)M * Nkv);    //  4 MB
    unsigned short* vb  = take((size_t)M * Nkv);    //  4 MB
    unsigned short* ob  = take((size_t)M * Cc);     // 16 MB   (~76 MB total)

    auto cast = [&](const float* src, unsigned short* dst, size_t n) {
        int n4 = (int)(n / 4);
        cast_bf16_kernel<<<dim3((n4 + 255) / 256), dim3(256), 0, stream>>>(src, dst, n4);
    };
    cast(x,  xb,  (size_t)M * Cc);
    cast(wq, wqb, (size_t)Cc * Cc);
    cast(wk, wkb, (size_t)Nkv * Cc);
    cast(wv, wvb, (size_t)Nkv * Cc);
    cast(wo, wob, (size_t)Cc * Cc);

    dim3 blk(256);
    gemm_wmma_kernel<true><<<dim3(Cc / 128, M / 128), blk, 0, stream>>>(
        xb, wqb, qb, M, Cc, Cc);
    gemm_wmma_kernel<true><<<dim3(Nkv / 128, M / 128), blk, 0, stream>>>(
        xb, wkb, kb, M, Nkv, Cc);
    gemm_wmma_kernel<true><<<dim3(Nkv / 128, M / 128), blk, 0, stream>>>(
        xb, wvb, vb, M, Nkv, Cc);

    attn_wmma_kernel<<<dim3(T / 64, Hq, Bz), dim3(128), 0, stream>>>(qb, kb, vb, ob);

    gemm_wmma_kernel<false><<<dim3(Cc / 128, M / 128), blk, 0, stream>>>(
        ob, wob, out, M, Cc, Cc);
}
